// RNN_87110526697777
// MI455X (gfx1250) — compile-verified, matrix operand order checked by reading, not run on hardware
//
#include <hip/hip_runtime.h>
#include <hip/hip_bf16.h>
#include <math.h>

typedef __bf16 bf16_t;
typedef __attribute__((ext_vector_type(16))) __bf16 v16bf;
typedef __attribute__((ext_vector_type(8)))  float  v8f;
typedef unsigned int u32x4 __attribute__((ext_vector_type(4)));
typedef int          i32x8 __attribute__((ext_vector_type(8)));
typedef int          i32x4 __attribute__((ext_vector_type(4)));

#define B_   32
#define T_   512
#define H_   512
#define L_   2
#define K2_  1024   // 2*H
#define NWG_ 16     // workgroups per layer

// LDS layout: padded row pitch 1040 bf16 elems (2080B, row stride == 8 dwords mod 64
// banks -> 2-way instead of 16-way conflicts). Pad of 8 elems lands at K=512 boundary.
#define PITCH_   1040
#define HALF1_   520                 // 512 data + 8 pad elems
#define SWG_ELEMS (64 * PITCH_)      // gates slice: 64 cols
#define SWC_ELEMS (32 * PITCH_)      // candidate slice: 32 cols
#define SMEM_BYTES ((SWG_ELEMS + SWC_ELEMS) * 2)

#if defined(__has_builtin)
#if __has_builtin(__builtin_amdgcn_tensor_load_to_lds)
#define USE_TDM 1
#endif
#endif

// ---------- prep kernels ----------
__global__ void f32_to_bf16_kernel(const float* __restrict__ in, bf16_t* __restrict__ out, int n) {
  int i = blockIdx.x * blockDim.x + threadIdx.x;
  if (i < n) out[i] = (bf16_t)in[i];
}

// in: [L][rows][cols] f32 (row = K-in, col = N-out)  ->  out: [L][cols][rows] bf16 (N-major)
__global__ void transpose_bf16_kernel(const float* __restrict__ in, bf16_t* __restrict__ out,
                                      int rows, int cols) {
  int l = blockIdx.y;
  int i = blockIdx.x * blockDim.x + threadIdx.x;
  if (i < rows * cols) {
    int r = i / cols, c = i % cols;
    out[(size_t)l * rows * cols + (size_t)c * rows + r] =
        (bf16_t)in[(size_t)l * rows * cols + i];
  }
}

// ---------- cross-workgroup barrier (per layer, monotonic counter) ----------
__device__ __forceinline__ void layer_barrier(unsigned* ctr, unsigned target) {
  __syncthreads();
  if (threadIdx.x == 0) {
    __threadfence();
    atomicAdd(ctr, 1u);
    while (__hip_atomic_load(ctr, __ATOMIC_ACQUIRE, __HIP_MEMORY_SCOPE_AGENT) < target)
      __builtin_amdgcn_s_sleep(1);
  }
  __syncthreads();
}

// ---------- B-fragment load from LDS (16B-aligned halves -> two ds_load_b128) ----------
__device__ __forceinline__ v16bf lds_frag(const bf16_t* p) {
  union { uint4 u[2]; v16bf v; } t;
  t.u[0] = *(const uint4*)(p);
  t.u[1] = *(const uint4*)(p + 8);
  return t.v;
}

#ifdef USE_TDM
// TDM load of a [rows x 1024] bf16 tile (row stride 1024 elems) into LDS at lds_off,
// with pad_enable: +4 dwords after every 256 dwords  ->  1040-elem LDS row pitch.
__device__ __forceinline__ void tdm_load_tile(unsigned lds_off, const void* gsrc, unsigned rows) {
  unsigned long long ga = (unsigned long long)(uintptr_t)gsrc;
  u32x4 g0 = { 1u,                                   // count=1 (valid descriptor)
               lds_off,                               // lds_addr (bytes)
               (unsigned)(ga & 0xffffffffu),          // global_addr[31:0]
               (unsigned)((ga >> 32) & 0x01ffffffu)   // global_addr[56:32]
                 | 0x80000000u };                     // type=2 ("image")
  i32x8 g1 = {
    (int)0x07D10000,        // data_size=1(2B) | pad_enable | pad_interval=7(256dw) | pad_amount=3(4dw)
    (int)(1024u << 16),     // tensor_dim0[15:0]=1024 in bits[31:16]
    (int)(rows  << 16),     // tensor_dim0[31:16]=0 | tensor_dim1[15:0]=rows
    (int)(1024u << 16),     // tensor_dim1[31:16]=0 | tile_dim0=1024
    (int)rows,              // tile_dim1=rows, tile_dim2=0
    (int)1024,              // tensor_dim0_stride[31:0]=1024 elems
    0, 0 };
  i32x4 z4 = { 0, 0, 0, 0 };
#if __clang_major__ >= 23
  i32x8 z8 = { 0, 0, 0, 0, 0, 0, 0, 0 };
  __builtin_amdgcn_tensor_load_to_lds(g0, g1, z4, z4, z8, 0);
#else
  __builtin_amdgcn_tensor_load_to_lds(g0, g1, z4, z4, 0);
#endif
}
#endif

// ---------- persistent GRU kernel: 32 WGs = 2 layers x 16 column-slices ----------
__global__ __launch_bounds__(256, 1) void gru_persistent_kernel(
    const bf16_t* __restrict__ xbf,      // [B][T][H] bf16
    const int*    __restrict__ seq_lens, // [B]
    const bf16_t* __restrict__ WTg,      // [L][2H(out)][2H(in)] bf16
    const float*  __restrict__ bg,       // [L][2H]
    const bf16_t* __restrict__ WTc,      // [L][H(out)][2H(in)] bf16
    const float*  __restrict__ bc,       // [L][H]
    bf16_t* __restrict__ y0,             // [T][B][H] bf16, layer0 output
    bf16_t* __restrict__ hbuf,           // [L][B][H] bf16 hidden state
    bf16_t* __restrict__ rhbuf,          // [L][B][H] bf16 (r*h)
    float*  __restrict__ ubuf,           // [L][B][H] f32 update gate
    unsigned* __restrict__ readyCnt,     // [T] layer0->layer1 ready counters
    unsigned* __restrict__ bar,          // [L] barrier counters
    float* __restrict__ out)             // [B*T][H] final output
{
  extern __shared__ char smem[];
  bf16_t* sWg = (bf16_t*)smem;                        // [64][1040]
  bf16_t* sWc = (bf16_t*)(smem + SWG_ELEMS * 2);      // [32][1040]

  const int layer = blockIdx.x >> 4;   // 0..1
  const int g     = blockIdx.x & 15;   // column-slice id within layer
  const int tid   = threadIdx.x;
  const int wave  = tid >> 5;          // 0..7 (wave32)
  const int lane  = tid & 31;
  const int lan16 = lane & 15;
  const int koffsel = (lane >> 4) << 4;   // lanes 16..31 hold K+16..K+31
  const int rowsel  = (lane >> 4) << 3;   // C/D: lanes 16..31 hold rows M+8..M+15

  // ---- stage this WG's weight slices into LDS (once; weights are step-invariant) ----
  const bf16_t* gWg = WTg + ((size_t)layer * K2_ + g * 64) * K2_;   // 64 rows x 1024
  const bf16_t* gWc = WTc + ((size_t)layer * H_  + g * 32) * K2_;   // 32 rows x 1024
#ifdef USE_TDM
  if (wave == 0) {
    tdm_load_tile(0u, gWg, 64u);
    tdm_load_tile((unsigned)(SWG_ELEMS * 2), gWc, 32u);
#if __has_builtin(__builtin_amdgcn_s_wait_tensorcnt)
    __builtin_amdgcn_s_wait_tensorcnt(0);
#endif
  }
#else
  for (int j = tid; j < 64 * 128; j += 256) {         // 8-elem (16B) chunks
    int row = j >> 7, k8 = (j & 127) * 8;
    *(uint4*)(sWg + row * PITCH_ + k8 + ((k8 >= 512) ? 8 : 0)) =
        *(const uint4*)(gWg + row * K2_ + k8);
  }
  for (int j = tid; j < 32 * 128; j += 256) {
    int row = j >> 7, k8 = (j & 127) * 8;
    *(uint4*)(sWc + row * PITCH_ + k8 + ((k8 >= 512) ? 8 : 0)) =
        *(const uint4*)(gWc + row * K2_ + k8);
  }
#endif
  __syncthreads();

  unsigned gen = 0;

  for (int t = 0; t < T_; ++t) {
    // layer 1 waits for layer 0's y[t] (all 16 producer WGs)
    if (layer == 1) {
      if (tid == 0) {
        while (__hip_atomic_load(&readyCnt[t], __ATOMIC_ACQUIRE, __HIP_MEMORY_SCOPE_AGENT) < NWG_)
          __builtin_amdgcn_s_sleep(1);
      }
      __syncthreads();
    }

    // ================= Phase 1: gates = sigmoid([x_t, h] @ Wg + bg) =================
    {
      const int n0 = g * 64 + (wave & 3) * 16;   // gate column tile base (0..1008)
      const int m0 = (wave >> 2) * 16;           // batch-row tile base (0 or 16)
      const int mrow = m0 + lan16;

      const bf16_t* arow = (layer == 0) ? (xbf + ((size_t)mrow * T_ + t) * H_)
                                        : (y0  + ((size_t)t * B_ + mrow) * H_);
      const bf16_t* hrow = hbuf + ((size_t)layer * B_ + mrow) * H_;
      const bf16_t* brow = sWg + (size_t)((wave & 3) * 16 + lan16) * PITCH_;

      v8f acc = {};
      #pragma unroll 4
      for (int kc = 0; kc < 16; ++kc) {          // K = 0..511  (x_t half)
        const int kk = kc * 32 + koffsel;
        v16bf av = *(const v16bf*)(arow + kk);
        v16bf bv = lds_frag(brow + kk);
        acc = __builtin_amdgcn_wmma_f32_16x16x32_bf16(false, av, false, bv,
                                                      (short)0, acc, false, false);
      }
      #pragma unroll 4
      for (int kc = 0; kc < 16; ++kc) {          // K = 512..1023 (h half)
        const int kk = kc * 32 + koffsel;
        v16bf av = *(const v16bf*)(hrow + kk);
        v16bf bv = lds_frag(brow + HALF1_ + kk);
        acc = __builtin_amdgcn_wmma_f32_16x16x32_bf16(false, av, false, bv,
                                                      (short)0, acc, false, false);
      }

      const int col   = n0 + lan16;              // gate column (0..1023)
      const float bias = bg[layer * K2_ + col];
      #pragma unroll
      for (int i = 0; i < 8; ++i) {
        const int row = m0 + i + rowsel;
        const float gte = 1.0f / (1.0f + __expf(-(acc[i] + bias)));
        const size_t hbase = ((size_t)layer * B_ + row) * H_;
        if (col < H_) {                          // r gate -> store r*h (bf16)
          const float hv = (float)hbuf[hbase + col];
          rhbuf[hbase + col] = (bf16_t)(gte * hv);
        } else {                                 // u gate -> store f32
          ubuf[hbase + (col - H_)] = gte;
        }
      }
    }

    ++gen;
    layer_barrier(&bar[layer], gen * NWG_);      // all r*h / u visible

    // ============ Phase 2: c = tanh([x_t, r*h] @ Wc + bc); h update ============
    if (wave < 4) {
      const int n0 = g * 32 + (wave & 1) * 16;   // candidate column tile (0..496)
      const int m0 = (wave >> 1) * 16;
      const int mrow = m0 + lan16;

      const bf16_t* arow  = (layer == 0) ? (xbf + ((size_t)mrow * T_ + t) * H_)
                                         : (y0  + ((size_t)t * B_ + mrow) * H_);
      const bf16_t* rhrow = rhbuf + ((size_t)layer * B_ + mrow) * H_;
      const bf16_t* brow  = sWc + (size_t)((wave & 1) * 16 + lan16) * PITCH_;

      v8f acc = {};
      #pragma unroll 4
      for (int kc = 0; kc < 16; ++kc) {          // x_t half
        const int kk = kc * 32 + koffsel;
        v16bf av = *(const v16bf*)(arow + kk);
        v16bf bv = lds_frag(brow + kk);
        acc = __builtin_amdgcn_wmma_f32_16x16x32_bf16(false, av, false, bv,
                                                      (short)0, acc, false, false);
      }
      #pragma unroll 4
      for (int kc = 0; kc < 16; ++kc) {          // r*h half
        const int kk = kc * 32 + koffsel;
        v16bf av = *(const v16bf*)(rhrow + kk);
        v16bf bv = lds_frag(brow + HALF1_ + kk);
        acc = __builtin_amdgcn_wmma_f32_16x16x32_bf16(false, av, false, bv,
                                                      (short)0, acc, false, false);
      }

      const int col   = n0 + lan16;              // hidden column (0..511)
      const float bias = bc[layer * H_ + col];
      #pragma unroll
      for (int i = 0; i < 8; ++i) {
        const int row = m0 + i + rowsel;
        const float c = tanhf(acc[i] + bias);
        const size_t hidx = ((size_t)layer * B_ + row) * H_ + col;
        const float u    = ubuf[hidx];
        const float hold = (float)hbuf[hidx];
        const float hnew = u * hold + (1.0f - u) * c;
        const bool  msk  = t < seq_lens[row];
        hbuf[hidx] = (bf16_t)(msk ? hnew : hold);
        const float y = msk ? hnew : 0.0f;
        if (layer == 0) y0[((size_t)t * B_ + row) * H_ + col] = (bf16_t)y;
        else            out[((size_t)row * T_ + t) * H_ + col] = y;
      }
    }
    __syncthreads();
    if (layer == 0 && tid == 0) {                // publish y[t] slice
      __threadfence();
      atomicAdd(&readyCnt[t], 1u);
    }

    ++gen;
    layer_barrier(&bar[layer], gen * NWG_);      // h consumed before next step writes
  }
}

// ---------- host launcher ----------
extern "C" void kernel_launch(void* const* d_in, const int* in_sizes, int n_in,
                              void* d_out, int out_size, void* d_ws, size_t ws_size,
                              hipStream_t stream) {
  (void)in_sizes; (void)n_in; (void)out_size; (void)ws_size;
  const float* x   = (const float*)d_in[0];
  const int*   seq = (const int*)  d_in[1];
  const float* Wg  = (const float*)d_in[2];
  const float* bg  = (const float*)d_in[3];
  const float* Wc  = (const float*)d_in[4];
  const float* bc  = (const float*)d_in[5];
  float* out = (float*)d_out;

  char* p = (char*)d_ws;
  auto carve = [&](size_t bytes) -> char* {
    char* r = p;
    p += (bytes + 255) & ~(size_t)255;
    return r;
  };
  bf16_t*  WTg      = (bf16_t*) carve((size_t)L_ * K2_ * K2_ * 2);
  bf16_t*  WTc      = (bf16_t*) carve((size_t)L_ * H_  * K2_ * 2);
  bf16_t*  xbf      = (bf16_t*) carve((size_t)B_ * T_  * H_  * 2);
  bf16_t*  y0       = (bf16_t*) carve((size_t)T_ * B_  * H_  * 2);
  bf16_t*  hbuf     = (bf16_t*) carve((size_t)L_ * B_  * H_  * 2);
  bf16_t*  rhbuf    = (bf16_t*) carve((size_t)L_ * B_  * H_  * 2);
  float*   ubuf     = (float*)  carve((size_t)L_ * B_  * H_  * 4);
  unsigned* readyCnt= (unsigned*)carve((size_t)T_ * 4);
  unsigned* bar     = (unsigned*)carve(256);

  // zero mutable state each call (graph-capture safe, deterministic)
  hipMemsetAsync(hbuf,     0, (size_t)L_ * B_ * H_ * 2, stream);
  hipMemsetAsync(readyCnt, 0, (size_t)T_ * 4,           stream);
  hipMemsetAsync(bar,      0, 256,                      stream);

  // prep: bf16 convert + weight transposes (W^T[N][K], K-contiguous for B-fragments)
  const int n = B_ * T_ * H_;
  f32_to_bf16_kernel<<<dim3((n + 255) / 256), dim3(256), 0, stream>>>(x, xbf, n);
  transpose_bf16_kernel<<<dim3((K2_ * K2_ + 255) / 256, L_), dim3(256), 0, stream>>>(Wg, WTg, K2_, K2_);
  transpose_bf16_kernel<<<dim3((K2_ * H_  + 255) / 256, L_), dim3(256), 0, stream>>>(Wc, WTc, K2_, H_);

  // persistent pipelined GRU: 32 WGs (2 layers x 16 column slices), 8 wave32s each,
  // 199680B dynamic LDS per WG (weights resident; WGP has 320KB)
  gru_persistent_kernel<<<dim3(L_ * NWG_), dim3(256), SMEM_BYTES, stream>>>(
      xbf, seq, WTg, bg, WTc, bc, y0, hbuf, rhbuf, ubuf, readyCnt, bar, out);
}
